// PSPGnnDGL_9423158248223
// MI455X (gfx1250) — matrix operand order repaired and checked by reading, not compiled, not run on hardware
//
#include <hip/hip_runtime.h>

// ---------------------------------------------------------------------------
// Types for CDNA5 WMMA (wave32)
// ---------------------------------------------------------------------------
typedef __bf16 bf16_t;
typedef __attribute__((ext_vector_type(16))) __bf16 v16bf;
typedef __attribute__((ext_vector_type(8)))  float  v8f;
typedef __attribute__((ext_vector_type(4)))  unsigned int u32x4;
typedef __attribute__((ext_vector_type(4)))  float  f32x4;

union FragBF {
    u32x4 u[2];   // 2 x 16B chunks
    v16bf v;      // 16 bf16 = 32B per lane
};

__device__ __forceinline__ u32x4 zero_u4() {
    u32x4 z; z[0] = 0u; z[1] = 0u; z[2] = 0u; z[3] = 0u; return z;
}

__device__ __forceinline__ float gelu_f(float x) {
    // jax.nn.gelu (approximate=True, tanh form)
    const float c = 0.7978845608028654f;
    float x3 = x * x * x;
    return 0.5f * x * (1.0f + tanhf(c * (x + 0.044715f * x3)));
}

__device__ __forceinline__ void atomicMaxF(float* addr, float val) {
    int* ai = (int*)addr;
    int old = __float_as_int(*addr);
    while (__int_as_float(old) < val) {
        int assumed = old;
        old = atomicCAS(ai, assumed, __float_as_int(val));
        if (old == assumed) break;
    }
}

// ---------------------------------------------------------------------------
// WMMA GEMM core: one wave accumulates a 16x16 f32 tile over K (bf16 inputs).
// A-fragment lane layout (16-bit A 16x32): lane m%16 = row, lanes 0-15 hold
// K [0,8)+[16,24), lanes 16-31 hold K [8,16)+[24,32) relative to k0.
// B-fragment (16-bit B 32x16): lane n%16 = col, same K split as A (8+8 K per lane),
// K offset (lane/16)*8, second 8 at +16.  W pre-transposed to Wt[Nc,K] -> contiguous loads.
// GUARD=false is the block-uniform full-tile fast path (no EXEC juggling).
// ---------------------------------------------------------------------------
template<bool GUARD>
__device__ __forceinline__ v8f gemm_core(const bf16_t* __restrict__ X,
                                         const bf16_t* __restrict__ Wt,
                                         long arow, bool arowok, int col,
                                         int sel, int K)
{
    v8f acc = {};
    for (int k0 = 0; k0 < K; k0 += 32) {
        FragBF a, b;
        if (!GUARD || arowok) {
            const bf16_t* ap = X + arow * (long)K + k0 + sel * 8;
            a.u[0] = *(const u32x4*)(ap);
            a.u[1] = *(const u32x4*)(ap + 16);
        } else {
            a.u[0] = zero_u4(); a.u[1] = zero_u4();
        }
        const bf16_t* bp = Wt + (long)col * K + k0 + sel * 8;
        b.u[0] = *(const u32x4*)(bp);
        b.u[1] = *(const u32x4*)(bp + 16);
        acc = __builtin_amdgcn_wmma_f32_16x16x32_bf16(
                  false, a.v, false, b.v, (short)0, acc, false, false);
    }
    return acc;
}

template<bool GUARD>
__device__ __forceinline__ void gemm_epilogue(v8f acc, long row0, int sel, int col,
                                              const float* __restrict__ bias,
                                              const float* resid,
                                              float* Yf, int ldYf,
                                              bf16_t* __restrict__ Yb, int ldYb,
                                              long M, int Nc, int act)
{
    const float bv = bias ? bias[col] : 0.0f;
#pragma unroll
    for (int r = 0; r < 8; ++r) {
        const long row = row0 + sel * 8 + r;   // C layout: lanes16-31 hold M=8..15
        if (!GUARD || row < M) {
            float v = acc[r] + bv;
            if (act == 1) v = gelu_f(v);
            if (resid) v += resid[row * (long)Nc + col];
            if (Yf) Yf[row * (long)ldYf + col] = v;
            if (Yb) Yb[row * (long)ldYb + col] = (bf16_t)v;
        }
    }
}

// Y = act(X[M,K](bf16) @ Wt[Nc,K]^T + bias) (+resid). Block = (Nc/16) waves,
// each wave owns one 16x16 tile of a 16-row stripe.
__global__ void wmma_gemm_kernel(const bf16_t* __restrict__ X,
                                 const bf16_t* __restrict__ Wt,
                                 const float* __restrict__ bias,
                                 const float* resid,
                                 float* Yf, int ldYf,
                                 bf16_t* __restrict__ Yb, int ldYb,
                                 int M, int K, int Nc, int act)
{
    const int tid  = threadIdx.x;
    const int lane = tid & 31;
    const int wv   = tid >> 5;
    const int sel  = lane >> 4;      // 0 or 1 (half-wave)
    const int lcol = lane & 15;
    const int n0   = wv << 4;
    const long row0 = (long)blockIdx.x << 4;

    const long arow = row0 + lcol;
    const int  col  = n0 + lcol;
    const bool full = (row0 + 16 <= M);      // block-uniform

    if (full) {
        v8f acc = gemm_core<false>(X, Wt, arow, true, col, sel, K);
        gemm_epilogue<false>(acc, row0, sel, col, bias, resid,
                             Yf, ldYf, Yb, ldYb, M, Nc, act);
    } else {
        v8f acc = gemm_core<true>(X, Wt, arow, arow < M, col, sel, K);
        gemm_epilogue<true>(acc, row0, sel, col, bias, resid,
                            Yf, ldYf, Yb, ldYb, M, Nc, act);
    }
}

// ---------------------------------------------------------------------------
// Fused edge-attention logits: per 16-edge tile, WMMA ef@W_fij (K=64,Nc=128),
// gather s_ni[src]/s_nj[dst], leaky_relu, dot with attn per head (LDS atomic
// reduce), write e[E,2] and atomic-max into per-(dst,head) running max.
// Block = 8 waves (256 thr), wave w owns cols [16w,16w+16).
// ---------------------------------------------------------------------------
template<bool GUARD>
__device__ __forceinline__ void edge_accum(v8f acc, long e0, int sel, int n,
                                           const float* __restrict__ s_ni,
                                           const float* __restrict__ s_nj,
                                           const int* __restrict__ src,
                                           const int* __restrict__ dst,
                                           float av, int head,
                                           float* e_part, int E_)
{
#pragma unroll
    for (int r = 0; r < 8; ++r) {
        const int  m = sel * 8 + r;
        const long e = e0 + m;
        if (!GUARD || e < E_) {
            const int s = src[e];
            const int t = dst[e];
            float v = acc[r] + s_ni[(size_t)s * 128 + n] + s_nj[(size_t)t * 128 + n];
            float lv = v > 0.0f ? v : 0.2f * v;        // leaky_relu(0.2)
            atomicAdd(&e_part[m * 2 + head], lv * av); // LDS reduce over d
        }
    }
}

__global__ void edge_logits_kernel(const bf16_t* __restrict__ efb,   // [E,64]
                                   const bf16_t* __restrict__ Wfijt, // [128,64]
                                   const float* __restrict__ s_ni,   // [N,128]
                                   const float* __restrict__ s_nj,   // [N,128]
                                   const float* __restrict__ attn,   // [2,64]
                                   const int* __restrict__ src,
                                   const int* __restrict__ dst,
                                   float* __restrict__ e_out,        // [E,2]
                                   float* __restrict__ mmax,         // [N,2]
                                   int E_)
{
    __shared__ float e_part[32];   // [16 edges][2 heads]
    const int tid = threadIdx.x;
    if (tid < 32) e_part[tid] = 0.0f;
    __syncthreads();

    const int lane = tid & 31;
    const int wv   = tid >> 5;
    const int sel  = lane >> 4;
    const int lcol = lane & 15;
    const int n0   = wv << 4;
    const long e0  = (long)blockIdx.x << 4;

    const long arow = e0 + lcol;
    const int  n    = n0 + lcol;
    const int  head = n >> 6;
    const int  d    = n & 63;
    const float av  = attn[head * 64 + d];
    const bool full = (e0 + 16 <= E_);       // block-uniform

    if (full) {
        v8f acc = gemm_core<false>(efb, Wfijt, arow, true, n, sel, 64);
        edge_accum<false>(acc, e0, sel, n, s_ni, s_nj, src, dst, av, head,
                          e_part, E_);
    } else {
        v8f acc = gemm_core<true>(efb, Wfijt, arow, arow < E_, n, sel, 64);
        edge_accum<true>(acc, e0, sel, n, s_ni, s_nj, src, dst, av, head,
                         e_part, E_);
    }
    __syncthreads();

    if (tid < 32) {
        const int  m  = tid >> 1;
        const int  hh = tid & 1;
        const long e  = e0 + m;
        if (e < E_) {
            const float ev = e_part[tid];
            e_out[e * 2 + hh] = ev;
            atomicMaxF(&mmax[(size_t)dst[e] * 2 + hh], ev);  // segment_max
        }
    }
}

// w = exp(e - mmax[dst]); z[dst] += w  (in-place over e buffer)
__global__ void edge_softmax_kernel(float* __restrict__ e_w,
                                    const float* __restrict__ mmax,
                                    const int* __restrict__ dst,
                                    float* __restrict__ z, long total)
{
    long i = (long)blockIdx.x * blockDim.x + threadIdx.x;
    if (i >= total) return;
    long e = i >> 1; int h = (int)(i & 1);
    float w = __expf(e_w[i] - mmax[(size_t)dst[e] * 2 + h]);
    e_w[i] = w;
    atomicAdd(&z[(size_t)dst[e] * 2 + h], w);
}

// agg[dst] += hnode[src] * (w / z[dst]); thread handles 4 cols of one edge
__global__ void aggregate_kernel(const float* __restrict__ w,
                                 const float* __restrict__ z,
                                 const float* __restrict__ hnode,
                                 const int* __restrict__ src,
                                 const int* __restrict__ dst,
                                 float* __restrict__ agg, long total)
{
    long i = (long)blockIdx.x * blockDim.x + threadIdx.x;
    if (i >= total) return;
    long e = i >> 5; int q = (int)(i & 31);      // 4-col group, head = q/16
    int h = q >> 4;
    int t = dst[e];
    float a = w[e * 2 + h] / z[(size_t)t * 2 + h];
    const f32x4 hv = *(const f32x4*)&hnode[(size_t)src[e] * 128 + 4 * q];
    float* dp = &agg[(size_t)t * 128 + 4 * q];
    atomicAdd(dp + 0, hv[0] * a);
    atomicAdd(dp + 1, hv[1] * a);
    atomicAdd(dp + 2, hv[2] * a);
    atomicAdd(dp + 3, hv[3] * a);
}

// ef = type_emb[etype] + rid_emb[erid] + att_rc@rc_W + rc_b + att_rp@rp_W + rp_b
__global__ void edge_feat_kernel(const float* __restrict__ att_rc,
                                 const float* __restrict__ att_rp,
                                 const float* __restrict__ type_emb,
                                 const float* __restrict__ rid_emb,
                                 const float* __restrict__ rc_W,
                                 const float* __restrict__ rc_b,
                                 const float* __restrict__ rp_W,
                                 const float* __restrict__ rp_b,
                                 const int* __restrict__ etype,
                                 const int* __restrict__ erid,
                                 bf16_t* __restrict__ efb, long total)
{
    long i = (long)blockIdx.x * blockDim.x + threadIdx.x;
    if (i >= total) return;
    long e = i >> 6; int d = (int)(i & 63);
    float v = type_emb[etype[e] * 64 + d] + rid_emb[erid[e] * 64 + d]
            + rc_b[d] + rp_b[d];
    v += att_rc[e * 2 + 0] * rc_W[d] + att_rc[e * 2 + 1] * rc_W[64 + d];
    v += att_rp[e * 3 + 0] * rp_W[d] + att_rp[e * 3 + 1] * rp_W[64 + d]
       + att_rp[e * 3 + 2] * rp_W[128 + d];
    efb[i] = (bf16_t)v;
}

// W[K,Nc] f32 row-major -> Wt[Nc,K] bf16
__global__ void transpose_to_bf16(const float* __restrict__ W,
                                  bf16_t* __restrict__ Wt, int K, int Nc)
{
    int i = blockIdx.x * blockDim.x + threadIdx.x;
    if (i >= K * Nc) return;
    int c = i / K, k = i % K;
    Wt[i] = (bf16_t)W[(size_t)k * Nc + c];
}

__global__ void cvt_bf16_kernel(const float* __restrict__ s,
                                bf16_t* __restrict__ d, long n)
{
    long i = (long)blockIdx.x * blockDim.x + threadIdx.x;
    if (i < n) d[i] = (bf16_t)s[i];
}

__global__ void fill_kernel(float* __restrict__ p, float v, long n)
{
    long i = (long)blockIdx.x * blockDim.x + threadIdx.x;
    if (i < n) p[i] = v;
}

__global__ void copy_cols_kernel(const float* __restrict__ s, int lds_,
                                 float* __restrict__ d, int ldd, int coloff,
                                 int ncols, long M)
{
    long i = (long)blockIdx.x * blockDim.x + threadIdx.x;
    if (i >= M * ncols) return;
    long r = i / ncols; int c = (int)(i % ncols);
    d[r * ldd + coloff + c] = s[r * lds_ + c];
}

// column-wise max over 128-row chunks of out[:, 0:352], atomic-max into graph
__global__ void graph_max_kernel(const float* __restrict__ out,
                                 float* __restrict__ graph, int M)
{
    int c = threadIdx.x;
    if (c >= 352) return;
    long r0 = (long)blockIdx.x * 128;
    float m = -3.402823466e38f;
    for (int i = 0; i < 128; ++i) {
        long r = r0 + i;
        if (r < M) m = fmaxf(m, out[r * 704 + c]);
    }
    atomicMaxF(&graph[c], m);
}

__global__ void broadcast_kernel(float* __restrict__ out,
                                 const float* __restrict__ graph, long total)
{
    long i = (long)blockIdx.x * blockDim.x + threadIdx.x;
    if (i >= total) return;
    long r = i / 352; int c = (int)(i % 352);
    out[r * 704 + 352 + c] = graph[c];
}

// ---------------------------------------------------------------------------
extern "C" void kernel_launch(void* const* d_in, const int* in_sizes, int n_in,
                              void* d_out, int out_size, void* d_ws, size_t ws_size,
                              hipStream_t stream)
{
    (void)in_sizes; (void)n_in; (void)out_size; (void)ws_size;

    const float* feat     = (const float*)d_in[0];
    const float* att_rc   = (const float*)d_in[1];
    const float* att_rp   = (const float*)d_in[2];
    const float* type_emb = (const float*)d_in[3];
    const float* rid_emb  = (const float*)d_in[4];
    const float* rc_W     = (const float*)d_in[5];
    const float* rc_b     = (const float*)d_in[6];
    const float* rp_W     = (const float*)d_in[7];
    const float* rp_b     = (const float*)d_in[8];
    const float* fe_W1    = (const float*)d_in[9];
    const float* fe_b1    = (const float*)d_in[10];
    const float* fe_W2    = (const float*)d_in[11];
    const float* fe_b2    = (const float*)d_in[12];
    const float* W_ni     = (const float*)d_in[13];
    const float* W_nj     = (const float*)d_in[14];
    const float* W_fij    = (const float*)d_in[15];
    const float* W_node   = (const float*)d_in[16];
    const float* b_node   = (const float*)d_in[17];
    const float* attn     = (const float*)d_in[18];
    const float* mlp_W1   = (const float*)d_in[19];
    const float* mlp_b1   = (const float*)d_in[20];
    const float* mlp_W2   = (const float*)d_in[21];
    const float* mlp_b2   = (const float*)d_in[22];
    const int*   src      = (const int*)d_in[23];
    const int*   dst      = (const int*)d_in[24];
    const int*   etype    = (const int*)d_in[25];
    const int*   erid     = (const int*)d_in[26];
    float* out = (float*)d_out;

    constexpr int N = 50000, E = 640000, IN = 32, HID = 64, L = 4, F = 128;
    constexpr float NEG_INF = -3.402823466e38f;

    char* ws = (char*)d_ws;
    size_t off = 0;
    auto alloc = [&](size_t bytes) -> char* {
        char* p = ws + off;
        off += (bytes + 255) & ~(size_t)255;
        return p;
    };

    bf16_t* featb = (bf16_t*)alloc((size_t)N * IN * 2);
    bf16_t* efb   = (bf16_t*)alloc((size_t)E * HID * 2);
    bf16_t* hb    = (bf16_t*)alloc((size_t)N * HID * 2);
    float*  hf    = (float*) alloc((size_t)N * HID * 4);
    bf16_t* t1b   = (bf16_t*)alloc((size_t)N * HID * 2);
    float*  s_ni  = (float*) alloc((size_t)N * F * 4);
    float*  s_nj  = (float*) alloc((size_t)N * F * 4);
    float*  hnode = (float*) alloc((size_t)N * F * 4);
    float*  e_w   = (float*) alloc((size_t)E * 2 * 4);
    float*  mmax  = (float*) alloc((size_t)N * 2 * 4);
    float*  zbuf  = (float*) alloc((size_t)N * 2 * 4);
    float*  agg   = (float*) alloc((size_t)N * F * 4);
    bf16_t* aggb  = (bf16_t*)alloc((size_t)N * F * 2);
    bf16_t* t2b   = (bf16_t*)alloc((size_t)N * F * 2);
    float*  graph = (float*) alloc(352 * 4);

    bf16_t* fe_W1t = (bf16_t*)alloc((size_t)IN * HID * 2);
    bf16_t* fe_W2t = (bf16_t*)alloc((size_t)HID * HID * 2);
    bf16_t *Wnit[L], *Wnjt[L], *Wfijt[L], *Wnodet[L], *mW1t[L], *mW2t[L];
    for (int l = 0; l < L; ++l) {
        Wnit[l]   = (bf16_t*)alloc((size_t)HID * F * 2);
        Wnjt[l]   = (bf16_t*)alloc((size_t)HID * F * 2);
        Wfijt[l]  = (bf16_t*)alloc((size_t)HID * F * 2);
        Wnodet[l] = (bf16_t*)alloc((size_t)HID * F * 2);
        mW1t[l]   = (bf16_t*)alloc((size_t)F * F * 2);
        mW2t[l]   = (bf16_t*)alloc((size_t)F * HID * 2);
    }

    auto tpose = [&](const float* W, bf16_t* Wt, int K, int Nc) {
        int tot = K * Nc;
        transpose_to_bf16<<<(tot + 255) / 256, 256, 0, stream>>>(W, Wt, K, Nc);
    };
    auto gemm = [&](const bf16_t* X, const bf16_t* Wt, const float* bias,
                    const float* resid, float* Yf, int ldYf,
                    bf16_t* Yb, int ldYb, int M, int K, int Nc, int act) {
        dim3 grid((M + 15) / 16);
        dim3 block((Nc / 16) * 32);
        wmma_gemm_kernel<<<grid, block, 0, stream>>>(
            X, Wt, bias, resid, Yf, ldYf, Yb, ldYb, M, K, Nc, act);
    };
    auto fill = [&](float* p, float v, long n) {
        fill_kernel<<<(unsigned)((n + 255) / 256), 256, 0, stream>>>(p, v, n);
    };
    auto ccopy = [&](const float* s, int lds_, int coloff, int ncols) {
        long tot = (long)N * ncols;
        copy_cols_kernel<<<(unsigned)((tot + 255) / 256), 256, 0, stream>>>(
            s, lds_, out, 704, coloff, ncols, (long)N);
    };

    // --- one-time weight prep (tiny; stays resident in L2) ---
    tpose(fe_W1, fe_W1t, IN, HID);
    tpose(fe_W2, fe_W2t, HID, HID);
    for (int l = 0; l < L; ++l) {
        tpose(W_ni   + (size_t)l * HID * F, Wnit[l],   HID, F);
        tpose(W_nj   + (size_t)l * HID * F, Wnjt[l],   HID, F);
        tpose(W_fij  + (size_t)l * HID * F, Wfijt[l],  HID, F);
        tpose(W_node + (size_t)l * HID * F, Wnodet[l], HID, F);
        tpose(mlp_W1 + (size_t)l * F * F,   mW1t[l],   F, F);
        tpose(mlp_W2 + (size_t)l * F * HID, mW2t[l],   F, HID);
    }

    // --- inputs ---
    cvt_bf16_kernel<<<(unsigned)(((long)N * IN + 255) / 256), 256, 0, stream>>>(
        feat, featb, (long)N * IN);
    ccopy(feat, IN, 0, IN);                                 // feats[0]
    {
        long tot = (long)E * HID;
        edge_feat_kernel<<<(unsigned)((tot + 255) / 256), 256, 0, stream>>>(
            att_rc, att_rp, type_emb, rid_emb, rc_W, rc_b, rp_W, rp_b,
            etype, erid, efb, tot);
    }

    // --- feature encoder: h = gelu(feat@W1+b1)@W2+b2 ---
    gemm(featb, fe_W1t, fe_b1, nullptr, nullptr, 0, t1b, HID, N, IN, HID, 1);
    gemm(t1b, fe_W2t, fe_b2, nullptr, hf, HID, hb, HID, N, HID, HID, 0);
    ccopy(hf, HID, IN, HID);                                // feats[1]

    // --- EGAT layers ---
    for (int l = 0; l < L; ++l) {
        fill(mmax, NEG_INF, (long)N * 2);
        fill(zbuf, 0.0f, (long)N * 2);
        fill(agg, 0.0f, (long)N * F);

        gemm(hb, Wnit[l],   nullptr,          nullptr, s_ni,  F, nullptr, 0, N, HID, F, 0);
        gemm(hb, Wnjt[l],   nullptr,          nullptr, s_nj,  F, nullptr, 0, N, HID, F, 0);
        gemm(hb, Wnodet[l], b_node + (size_t)l * F, nullptr, hnode, F, nullptr, 0, N, HID, F, 0);

        edge_logits_kernel<<<(E + 15) / 16, 256, 0, stream>>>(
            efb, Wfijt[l], s_ni, s_nj, attn + (size_t)l * F, src, dst,
            e_w, mmax, E);

        {
            long tot = (long)E * 2;
            edge_softmax_kernel<<<(unsigned)((tot + 255) / 256), 256, 0, stream>>>(
                e_w, mmax, dst, zbuf, tot);
        }
        {
            long tot = (long)E * 32;
            aggregate_kernel<<<(unsigned)((tot + 255) / 256), 256, 0, stream>>>(
                e_w, zbuf, hnode, src, dst, agg, tot);
        }

        cvt_bf16_kernel<<<(unsigned)(((long)N * F + 255) / 256), 256, 0, stream>>>(
            agg, aggb, (long)N * F);
        gemm(aggb, mW1t[l], mlp_b1 + (size_t)l * F, nullptr,
             nullptr, 0, t2b, F, N, F, F, 1);
        gemm(t2b, mW2t[l], mlp_b2 + (size_t)l * HID, hf,       // residual +h
             hf, HID, hb, HID, N, F, HID, 0);
        ccopy(hf, HID, 96 + 64 * l, HID);                       // feats[2+l]
    }

    // --- graph max pooling + broadcast ---
    fill(graph, NEG_INF, 352);
    graph_max_kernel<<<(N + 127) / 128, 384, 0, stream>>>(out, graph, N);
    {
        long tot = (long)N * 352;
        broadcast_kernel<<<(unsigned)((tot + 255) / 256), 256, 0, stream>>>(
            out, graph, tot);
    }
}
